// ScaledDotProductAttention_10453950398993
// MI455X (gfx1250) — matrix operand order repaired and checked by reading, not compiled, not run on hardware
//
#include <hip/hip_runtime.h>

// ---------------------------------------------------------------------------
// Flash attention (B=16, N=4096, D=64) fp32 in/out on CDNA5 (gfx1250, wave32).
// bf16 WMMA (v_wmma_f32_16x16x32_bf16) for QK^T, PV, and row-sum (P·1);
// fp32 online softmax in exp2 domain. Double-buffered LDS, v_perm bf16 packing,
// permuted P columns so P staging stores are b64.
// ---------------------------------------------------------------------------

typedef __attribute__((ext_vector_type(16))) __bf16 bf16x16;
typedef __attribute__((ext_vector_type(8)))  float  f32x8;
typedef __attribute__((ext_vector_type(4)))  int    i32x4;

#define N_SEQ 4096
#define DHEAD 64
#define BATCH 16
#define WAVES 4        // waves per workgroup
#define TQ    32       // query rows per wave (2 x 16-row subtiles)
#define TK    64       // key tile (K-dim of PV GEMM = 2 x 32)
#define LPAD  8        // row pad: 64+8=72 ushorts = 144B = 9*16B (16B-aligned rows)

// scale = 1/sqrt(64) * log2(e): softmax done with exp2 (bare v_exp_f32)
#define QSCALE 0.1803368801111137f

// bare v_exp_f32 (exp2)
#define EXP2F(x) __builtin_amdgcn_exp2f(x)

union BFrag {
    bf16x16        v;
    unsigned short u[16];
};
struct V32 { i32x4 a, b; };   // 32 bytes

// raw v_max_num_f32 (skip fmaxf's sNaN canonicalization of shuffle results)
__device__ __forceinline__ float fmax_raw(float a, float b) {
    float d;
    asm("v_max_num_f32 %0, %1, %2" : "=v"(d) : "v"(a), "v"(b));
    return d;
}
// pack two f32 -> two bf16 (round-half-up) in one v_perm_b32
__device__ __forceinline__ unsigned pk2(float a, float b) {
    unsigned ua = __float_as_uint(a) + 0x8000u;
    unsigned ub = __float_as_uint(b) + 0x8000u;
    return __builtin_amdgcn_perm(ub, ua, 0x07060302u);
}
__device__ __forceinline__ unsigned short pk1(float a) {
    return (unsigned short)((__float_as_uint(a) + 0x8000u) >> 16);
}
// 16 contiguous bf16 (two ds_load_b128)
__device__ __forceinline__ bf16x16 ld_contig16(const unsigned short* p) {
    V32 t;
    t.a = *(const i32x4*)p;
    t.b = *(const i32x4*)(p + 8);
    return __builtin_bit_cast(bf16x16, t);
}
// A-fragment halves: 8 bf16 at p, 8 bf16 at p+16 (two ds_load_b128)
__device__ __forceinline__ bf16x16 ld_split16(const unsigned short* p) {
    V32 t;
    t.a = *(const i32x4*)p;
    t.b = *(const i32x4*)(p + 16);
    return __builtin_bit_cast(bf16x16, t);
}

__global__ __launch_bounds__(WAVES * 32)
void fa_fwd_bf16_wmma(const float* __restrict__ Q,
                      const float* __restrict__ K,
                      const float* __restrict__ V,
                      float* __restrict__ Out)
{
    __shared__ __align__(16) unsigned short Kt[2][TK][DHEAD + LPAD];   // 18KB
    __shared__ __align__(16) unsigned short Vt[2][TK][DHEAD + LPAD];   // 18KB
    __shared__ __align__(16) unsigned short Pt[WAVES][TQ][TK + LPAD];  // 18KB

    const int tid  = threadIdx.x;
    const int wave = tid >> 5;
    const int lane = tid & 31;
    const int ln   = lane & 15;
    const int hi   = lane >> 4;

    const int qtiles = N_SEQ / (WAVES * TQ);                 // 32
    const int b      = blockIdx.x / qtiles;
    const int qbase  = (blockIdx.x % qtiles) * (WAVES * TQ) + wave * TQ;

    const float* Qb = Q + (size_t)b * N_SEQ * DHEAD;
    const float* Kb = K + (size_t)b * N_SEQ * DHEAD;
    const float* Vb = V + (size_t)b * N_SEQ * DHEAD;

    // ---- Q as bf16 A-fragments [qt][d-half], scale*log2e folded in
    BFrag qa[2][2];
#pragma unroll
    for (int qt = 0; qt < 2; ++qt) {
        const float* qp = Qb + (size_t)(qbase + qt * 16 + ln) * DHEAD;
#pragma unroll
        for (int i = 0; i < 8; ++i) {
            const int kb = ((i < 4) ? 2 * i : 16 + 2 * (i - 4)) + (hi ? 8 : 0);
#pragma unroll
            for (int j = 0; j < 2; ++j) {
                qa[qt][0].u[2 * i + j] = pk1(qp[kb + j] * QSCALE);
                qa[qt][1].u[2 * i + j] = pk1(qp[32 + kb + j] * QSCALE);
            }
        }
    }

    // ---- constant ones B-fragment (for row-sum WMMA)
    BFrag onesf;
#pragma unroll
    for (int i = 0; i < 16; ++i) onesf.u[i] = 0x3F80;   // bf16 1.0

    // ---- state
    f32x8 o[2][4];
#pragma unroll
    for (int qt = 0; qt < 2; ++qt)
#pragma unroll
        for (int j = 0; j < 4; ++j)
            o[qt][j] = (f32x8){0.f,0.f,0.f,0.f,0.f,0.f,0.f,0.f};
    float mrow[2][8], lrow[2][8];
#pragma unroll
    for (int qt = 0; qt < 2; ++qt)
#pragma unroll
        for (int r = 0; r < 8; ++r) { mrow[qt][r] = -1e30f; lrow[qt][r] = 0.f; }

    // ---- staging: 64x64 f32 tile -> bf16 LDS (8 x b128 loads, 8 x b64 stores)
    auto stage = [&](const float* ksrc, const float* vsrc,
                     unsigned short (*kt)[DHEAD + LPAD],
                     unsigned short (*vt)[DHEAD + LPAD]) {
#pragma unroll
        for (int i = 0; i < 8; ++i) {
            const int e   = (tid + i * (WAVES * 32)) * 4;   // 0..4092 step 4
            const int row = e >> 6;
            const int col = e & 63;
            const float4 kq = *(const float4*)(ksrc + e);
            uint2 kw; kw.x = pk2(kq.x, kq.y); kw.y = pk2(kq.z, kq.w);
            *(uint2*)&kt[row][col] = kw;
            const float4 vq = *(const float4*)(vsrc + e);
            uint2 vw; vw.x = pk2(vq.x, vq.y); vw.y = pk2(vq.z, vq.w);
            *(uint2*)&vt[row][col] = vw;
        }
    };

    const int NT = N_SEQ / TK;                               // 64 key tiles
    stage(Kb, Vb, Kt[0], Vt[0]);

    for (int t = 0; t < NT; ++t) {
        __syncthreads();                                     // tile t ready, buf t^1 free
        if (t + 1 < NT)
            stage(Kb + (size_t)(t + 1) * TK * DHEAD,
                  Vb + (size_t)(t + 1) * TK * DHEAD,
                  Kt[(t + 1) & 1], Vt[(t + 1) & 1]);
        if (t + 2 < NT) {                                    // global_prefetch_b8
            __builtin_prefetch((const char*)(Kb + (size_t)(t + 2) * TK * DHEAD) + tid * 128, 0, 1);
            __builtin_prefetch((const char*)(Vb + (size_t)(t + 2) * TK * DHEAD) + tid * 128, 0, 1);
        }

        const unsigned short (*kt)[DHEAD + LPAD] = Kt[t & 1];
        const unsigned short (*vt)[DHEAD + LPAD] = Vt[t & 1];

        // ---- S = Q.K^T, kg-outer so each K^T B-fragment pair is short-lived
        f32x8 sg[2][4];
#pragma unroll
        for (int kg = 0; kg < 4; ++kg) {
            BFrag b0, b1;
            b0.v = ld_split16(&kt[kg * 16 + ln][ 0 + hi * 8]);
            b1.v = ld_split16(&kt[kg * 16 + ln][32 + hi * 8]);
#pragma unroll
            for (int qt = 0; qt < 2; ++qt) {
                f32x8 acc = (f32x8){0.f,0.f,0.f,0.f,0.f,0.f,0.f,0.f};
                acc = __builtin_amdgcn_wmma_f32_16x16x32_bf16(false, qa[qt][0].v, false, b0.v, (short)0, acc, false, false);
                acc = __builtin_amdgcn_wmma_f32_16x16x32_bf16(false, qa[qt][1].v, false, b1.v, (short)0, acc, false, false);
                sg[qt][kg] = acc;
            }
        }

        // ---- online softmax (exp2 domain) + stage P with permuted columns:
        //      new col c = 4*ln + kg  (orig key 16*kg + ln)  -> b64 stores
#pragma unroll
        for (int qt = 0; qt < 2; ++qt) {
#pragma unroll
            for (int r = 0; r < 8; ++r) {
                float mx = fmaxf(fmaxf(sg[qt][0][r], sg[qt][1][r]),
                                 fmaxf(sg[qt][2][r], sg[qt][3][r]));
                mx = fmax_raw(mx, __shfl_xor(mx, 1, 32));
                mx = fmax_raw(mx, __shfl_xor(mx, 2, 32));
                mx = fmax_raw(mx, __shfl_xor(mx, 4, 32));
                mx = fmax_raw(mx, __shfl_xor(mx, 8, 32));
                const float mnew  = fmax_raw(mrow[qt][r], mx);
                const float alpha = EXP2F(mrow[qt][r] - mnew);
                mrow[qt][r] = mnew;
                lrow[qt][r] *= alpha;                        // L added after sum-WMMA
                const float p0 = EXP2F(sg[qt][0][r] - mnew);
                const float p1 = EXP2F(sg[qt][1][r] - mnew);
                const float p2 = EXP2F(sg[qt][2][r] - mnew);
                const float p3 = EXP2F(sg[qt][3][r] - mnew);
                const int m = qt * 16 + r + 8 * hi;          // C layout: row = r + 8*hi
                uint2 pw; pw.x = pk2(p0, p1); pw.y = pk2(p2, p3);
                *(uint2*)&Pt[wave][m][4 * ln] = pw;          // cols 4*ln .. 4*ln+3
                o[qt][0][r] *= alpha; o[qt][1][r] *= alpha;
                o[qt][2][r] *= alpha; o[qt][3][r] *= alpha;
            }
        }

        // ---- reload P as A-fragments [qt][key-half] (2 x ds_load_b128 each)
        BFrag pa[2][2];
#pragma unroll
        for (int qt = 0; qt < 2; ++qt)
#pragma unroll
            for (int h = 0; h < 2; ++h)
                pa[qt][h].v = ld_contig16(&Pt[wave][qt * 16 + ln][h * 32 + hi * 16]);

        // ---- row sums via WMMA: L = P . ones (permutation-invariant)
#pragma unroll
        for (int qt = 0; qt < 2; ++qt) {
            f32x8 L = (f32x8){0.f,0.f,0.f,0.f,0.f,0.f,0.f,0.f};
            L = __builtin_amdgcn_wmma_f32_16x16x32_bf16(false, pa[qt][0].v, false, onesf.v, (short)0, L, false, false);
            L = __builtin_amdgcn_wmma_f32_16x16x32_bf16(false, pa[qt][1].v, false, onesf.v, (short)0, L, false, false);
#pragma unroll
            for (int r = 0; r < 8; ++r) lrow[qt][r] += L[r];
        }

        // ---- O += P . V : V B-fragment rows follow the inverse permutation
        //      sigma^-1(c) = 16*(c&3) + (c>>2)
#pragma unroll
        for (int jc = 0; jc < 4; ++jc) {
            BFrag bv[2];
#pragma unroll
            for (int kh = 0; kh < 2; ++kh)
#pragma unroll
                for (int i = 0; i < 8; ++i)
#pragma unroll
                    for (int j = 0; j < 2; ++j) {
                        const int c   = kh * 32 + (hi ? 16 : 0) + 2 * i + j;  // new-K index
                        const int key = 16 * (c & 3) + (c >> 2);              // orig V row
                        bv[kh].u[2 * i + j] = vt[key][jc * 16 + ln];
                    }
#pragma unroll
            for (int qt = 0; qt < 2; ++qt) {
                o[qt][jc] = __builtin_amdgcn_wmma_f32_16x16x32_bf16(false, pa[qt][0].v, false, bv[0].v, (short)0, o[qt][jc], false, false);
                o[qt][jc] = __builtin_amdgcn_wmma_f32_16x16x32_bf16(false, pa[qt][1].v, false, bv[1].v, (short)0, o[qt][jc], false, false);
            }
        }
    }

    // ---- epilogue: normalize by row sums, store fp32 (coalesced in 16-lane groups)
    float* outb = Out + (size_t)b * N_SEQ * DHEAD;
#pragma unroll
    for (int qt = 0; qt < 2; ++qt)
#pragma unroll
        for (int r = 0; r < 8; ++r) {
            const float inv = 1.0f / lrow[qt][r];
            const int   m   = qbase + qt * 16 + r + 8 * hi;
            float* op = outb + (size_t)m * DHEAD + ln;
#pragma unroll
            for (int jc = 0; jc < 4; ++jc)
                op[jc * 16] = o[qt][jc][r] * inv;
        }
}

extern "C" void kernel_launch(void* const* d_in, const int* in_sizes, int n_in,
                              void* d_out, int out_size, void* d_ws, size_t ws_size,
                              hipStream_t stream) {
    (void)in_sizes; (void)n_in; (void)out_size; (void)d_ws; (void)ws_size;
    const float* q = (const float*)d_in[0];
    const float* k = (const float*)d_in[1];
    const float* v = (const float*)d_in[2];
    // d_in[3] = masking: faithful no-op in the reference.
    float* out = (float*)d_out;

    const int qtiles = N_SEQ / (WAVES * TQ);   // 32
    dim3 grid(BATCH * qtiles);                 // 512 workgroups
    dim3 block(WAVES * 32);                    // 4 waves (wave32)
    fa_fwd_bf16_wmma<<<grid, block, 0, stream>>>(q, k, v, out);
}
